// TTS_4071628997186
// MI455X (gfx1250) — compile-verified, matrix operand order checked by reading, not run on hardware
//
#include <hip/hip_runtime.h>

// ---------------------------------------------------------------------------
// Types / helpers
// ---------------------------------------------------------------------------
typedef __attribute__((ext_vector_type(16))) _Float16 v16h;
typedef __attribute__((ext_vector_type(8)))  _Float16 v8h;
typedef __attribute__((ext_vector_type(8)))  float    v8f;

#define ACT_NONE 0
#define ACT_RELU 1
#define ACT_SIG  2
#define ACT_TANH 3

// Fragment loader for 16-bit WMMA operands, matching CDNA5 16x32 (A) layout.
// Source is a row-major [rows x ld] f16 matrix; p0 points at (row0, k0).
// lane L holds row = row0 + (L%16); chunk0 at k0+(L/16)*8, chunk1 at +16.
__device__ __forceinline__ v16h ldfrag(const _Float16* p0, int ld) {
  const int lane = threadIdx.x & 31;
  const _Float16* p = p0 + (lane & 15) * ld + ((lane >> 4) << 3);
  v8h lo = *(const v8h*)p;
  v8h hi = *(const v8h*)(p + 16);
  v16h f;
#pragma unroll
  for (int i = 0; i < 8; ++i) { f[i] = lo[i]; f[i + 8] = hi[i]; }
  return f;
}

__device__ __forceinline__ v8f wmma16(v16h a, v16h b, v8f c) {
  return __builtin_amdgcn_wmma_f32_16x16x32_f16(false, a, false, b,
                                                (short)0, c, false, false);
}

// Pull L2-resident weights toward the WGP ahead of the next GEMM phase.
// Lowers to global_prefetch_b8 (one cacheline per issued prefetch).
__device__ __forceinline__ void prefetch_w(const void* w, int bytes) {
  const char* p = (const char*)w;
  for (int off = (int)threadIdx.x * 256; off < bytes; off += 256 * 256)
    __builtin_prefetch(p + off, 0, 0);
}

// Async global->LDS copy of one DWORD per lane (ASYNCcnt-tracked).
__device__ __forceinline__ void async_g2l_b32(void* lds_dst, const void* gsrc) {
  unsigned lds = (unsigned)(unsigned long long)lds_dst;  // addr[31:0] = LDS offset
  asm volatile("global_load_async_to_lds_b32 %0, %1, off"
               :: "v"(lds), "v"(gsrc) : "memory");
}
__device__ __forceinline__ void wait_asynccnt0() {
  asm volatile("s_wait_asynccnt 0x0" ::: "memory");
}

// ---------------------------------------------------------------------------
// Persistent-workgroup GEMM: C(32 x N) = X(32 x K, LDS, ld=576) * W^T(N x K)
// 8 waves, each wave computes full 16x16 tiles (2 M-tiles x N/16 N-tiles).
// ---------------------------------------------------------------------------
#define LDX 576

__device__ void wgemm(const _Float16* X, const _Float16* W, int ldw,
                      int N, int K, const float* bias, int act,
                      _Float16* dst, int ldd) {
  const int wave = threadIdx.x >> 5;
  const int lane = threadIdx.x & 31;
  const int ntiles = (N >> 4) * 2;
  for (int t = wave; t < ntiles; t += 8) {
    const int m0 = (t & 1) << 4;
    const int n0 = (t >> 1) << 4;
    v8f c = {};
    for (int k0 = 0; k0 < K; k0 += 32) {
      v16h a = ldfrag(X + m0 * LDX + k0, LDX);
      v16h b = ldfrag(W + n0 * ldw + k0, ldw);
      c = wmma16(a, b, c);
    }
    const int col = n0 + (lane & 15);
    const int r0  = m0 + ((lane >> 4) << 3);
    const float bv = bias ? bias[col] : 0.0f;
#pragma unroll
    for (int i = 0; i < 8; ++i) {
      float v = c[i] + bv;
      if (act == ACT_RELU)      v = v > 0.f ? v : 0.f;
      else if (act == ACT_SIG)  v = 1.f / (1.f + __expf(-v));
      else if (act == ACT_TANH) v = tanhf(v);
      dst[(r0 + i) * ldd + col] = (_Float16)v;
    }
  }
}

// ---------------------------------------------------------------------------
// Generic tiled WMMA GEMM: C(M x N) f32 = A(M x K) f16 * W^T(N x K) f16 + bias
// One 16x16 tile per wave.
// ---------------------------------------------------------------------------
__global__ __launch_bounds__(256) void gemm_g(const _Float16* __restrict__ A,
                                              int lda,
                                              const _Float16* __restrict__ B,
                                              int ldb,
                                              float* __restrict__ C, int ldc,
                                              int Mt, int Nt, int K,
                                              const float* __restrict__ bias,
                                              int Nvalid) {
  const int tile = blockIdx.x * 8 + (threadIdx.x >> 5);
  if (tile >= Mt * Nt) return;
  const int m0 = (tile % Mt) << 4;
  const int n0 = (tile / Mt) << 4;
  v8f c = {};
  for (int k0 = 0; k0 < K; k0 += 32) {
    v16h a = ldfrag(A + (size_t)m0 * lda + k0, lda);
    v16h b = ldfrag(B + (size_t)n0 * ldb + k0, ldb);
    c = wmma16(a, b, c);
  }
  const int lane = threadIdx.x & 31;
  const int col = n0 + (lane & 15);
  const int r0  = m0 + ((lane >> 4) << 3);
  if (col < Nvalid) {
    const float bv = bias ? bias[col] : 0.f;
#pragma unroll
    for (int i = 0; i < 8; ++i) C[(size_t)(r0 + i) * ldc + col] = c[i] + bv;
  }
}

// ---------------------------------------------------------------------------
// Weight convert+transpose: src (K x N) f32 row-major -> dst (Npad x Kpad) f16
// row n of dst = column n of src, zero padded.
// ---------------------------------------------------------------------------
__global__ void convert_w(const float* __restrict__ src,
                          _Float16* __restrict__ dst,
                          int K, int N, int Kpad, int Npad) {
  const int i = blockIdx.x * blockDim.x + threadIdx.x;
  if (i >= Npad * Kpad) return;
  const int n = i / Kpad, k = i % Kpad;
  dst[i] = (n < N && k < K) ? (_Float16)src[(size_t)k * N + n] : (_Float16)0.f;
}

__global__ void cast_f16(const float* __restrict__ s, _Float16* __restrict__ d,
                         int n) {
  const int i = blockIdx.x * blockDim.x + threadIdx.x;
  if (i < n) d[i] = (_Float16)s[i];
}

// mem_style = relu(style_token @ Ws_pre + bs_pre); keys_s = mem_style @ Wk_s
__global__ void style_pre(const float* st, const float* Wsp, const float* bsp,
                          const float* Wks, float* mem_style, float* keys_s) {
  const int n = threadIdx.x;
  if (n >= 10) return;
  float m[2];
#pragma unroll
  for (int d = 0; d < 2; ++d) {
    float v = bsp[d];
    for (int j = 0; j < 2; ++j) v += st[n * 2 + j] * Wsp[j * 2 + d];
    m[d] = v > 0.f ? v : 0.f;
    mem_style[n * 2 + d] = m[d];
  }
#pragma unroll
  for (int a = 0; a < 2; ++a) {
    float v = 0.f;
    for (int d = 0; d < 2; ++d) v += m[d] * Wks[d * 2 + a];
    keys_s[n * 2 + a] = v;
  }
}

// ---------------------------------------------------------------------------
// Persistent decoder: single workgroup, 256 threads (8 wave32), 100 steps.
// ---------------------------------------------------------------------------
struct DecArgs {
  const float *enc, *mel, *spk_embed, *v_att, *Wq_s, *v_s, *mem_style, *keys_s,
      *keys;
  const int *speaker, *inp_mask;
  const _Float16 *Wp1t, *Wp2t, *Wgat, *Wcat, *Wqt, *Wat, *Wg1t, *Wc1t, *Wg2t,
      *Wc2t, *Wot;
  const float *bp1, *bp2, *bga, *bca, *ba, *bg1, *bc1, *bg2, *bc2, *bo;
  float *out_mel, *out_alpha;
};

__global__ __launch_bounds__(256) void decoder(DecArgs P) {
  __shared__ __align__(16) _Float16 Xs[32 * 576];   // GEMM input (concat)
  __shared__ __align__(16) _Float16 Ys[32 * 256];   // GEMM outputs (<=256)
  __shared__ __align__(16) _Float16 Gs[32 * 512];   // GRU gates (r|u)
  __shared__ __align__(16) _Float16 hA[32 * 256];   // attention GRU state
  __shared__ __align__(16) _Float16 h1s[32 * 256];  // decoder GRU1 state
  __shared__ __align__(16) _Float16 h2s[32 * 256];  // decoder GRU2 state
  __shared__ __align__(16) _Float16 aco[32 * 256];  // aco_in, then y1
  float* ebuf   = (float*)Gs;  // 32x128 scores; Gs dead during attention
  float* mstage = (float*)Gs;  // prev-frame staging; Gs dead at step start

  const int tid = threadIdx.x;

  for (int i = tid; i < 32 * 256; i += 256) { hA[i] = (_Float16)0.f; h1s[i] = (_Float16)0.f; h2s[i] = (_Float16)0.f; }
  for (int i = tid; i < 32 * 32; i += 256) {          // zero pad cols 544..575
    int b = i >> 5, c = 544 + (i & 31);
    Xs[b * 576 + c] = (_Float16)0.f;
  }
  __syncthreads();

  for (int s = 0; s < 100; ++s) {
    // ---- prenet: prev frame (teacher-forced) -> Xs[:,0:128]
    if (s == 0) {
      for (int i = tid; i < 32 * 128; i += 256) {
        int b = i >> 7, d = i & 127;
        Xs[b * 576 + d] = (_Float16)0.f;
      }
    } else {
      // stage via async global->LDS (ASYNCcnt path), then convert to f16
      for (int i = tid; i < 32 * 128; i += 256) {
        int b = i >> 7, d = i & 127;
        async_g2l_b32(&mstage[i],
                      P.mel + (size_t)b * 64000 + (s * 5 - 1) * 128 + d);
      }
      wait_asynccnt0();
      __syncthreads();
      for (int i = tid; i < 32 * 128; i += 256) {
        int b = i >> 7, d = i & 127;
        Xs[b * 576 + d] = (_Float16)mstage[i];
      }
    }
    prefetch_w(P.Wp1t, 256 * 128 * 2);
    __syncthreads();
    wgemm(Xs, P.Wp1t, 128, 256, 128, P.bp1, ACT_RELU, Ys, 256);
    __syncthreads();
    for (int i = tid; i < 32 * 256; i += 256) { int b = i >> 8, d = i & 255; Xs[b * 576 + d] = Ys[b * 256 + d]; }
    prefetch_w(P.Wp2t, 128 * 256 * 2);
    __syncthreads();
    wgemm(Xs, P.Wp2t, 256, 128, 256, P.bp2, ACT_RELU, Ys, 256);
    __syncthreads();

    // ---- attention GRU gates: X = [p(128), spk(32), hA(256)]
    for (int i = tid; i < 32 * 128; i += 256) { int b = i >> 7, d = i & 127; Xs[b * 576 + d] = Ys[b * 256 + d]; }
    for (int i = tid; i < 32 * 32; i += 256) { int b = i >> 5, d = i & 31;
      Xs[b * 576 + 128 + d] = (_Float16)P.spk_embed[P.speaker[b] * 32 + d]; }
    for (int i = tid; i < 32 * 256; i += 256) { int b = i >> 8, d = i & 255; Xs[b * 576 + 160 + d] = hA[b * 256 + d]; }
    prefetch_w(P.Wgat, 512 * 416 * 2);
    __syncthreads();
    wgemm(Xs, P.Wgat, 416, 512, 416, P.bga, ACT_SIG, Gs, 512);
    __syncthreads();
    for (int i = tid; i < 32 * 256; i += 256) { int b = i >> 8, d = i & 255;
      Xs[b * 576 + 160 + d] = (_Float16)((float)Gs[b * 512 + d] * (float)hA[b * 256 + d]); }
    prefetch_w(P.Wcat, 256 * 416 * 2);
    __syncthreads();
    wgemm(Xs, P.Wcat, 416, 256, 416, P.bca, ACT_TANH, Ys, 256);
    __syncthreads();
    for (int i = tid; i < 32 * 256; i += 256) { int b = i >> 8, d = i & 255;
      float u = (float)Gs[b * 512 + 256 + d];
      float h = (float)hA[b * 256 + d], c = (float)Ys[b * 256 + d];
      hA[b * 256 + d] = (_Float16)(u * h + (1.f - u) * c); }
    __syncthreads();

    // ---- q @ Wq
    for (int i = tid; i < 32 * 256; i += 256) { int b = i >> 8, d = i & 255; Xs[b * 576 + d] = hA[b * 256 + d]; }
    prefetch_w(P.Wqt, 256 * 256 * 2);
    __syncthreads();
    wgemm(Xs, P.Wqt, 256, 256, 256, nullptr, ACT_NONE, Ys, 256);
    __syncthreads();

    // ---- scores e[b][t] = v . tanh(keys + qW)
    for (int i = tid; i < 32 * 128; i += 256) {
      int b = i >> 7, t = i & 127;
      const float* kp = P.keys + ((size_t)(b * 128 + t)) * 256;
      float acc = 0.f;
      for (int a = 0; a < 256; ++a)
        acc += P.v_att[a] * tanhf(kp[a] + (float)Ys[b * 256 + a]);
      ebuf[i] = acc;
    }
    __syncthreads();

    // ---- masked softmax + alpha output (one lane per batch row)
    if (tid < 32) {
      int b = tid, L = P.inp_mask[b];
      float mx = -1e30f;
      for (int t = 0; t < L; ++t) mx = fmaxf(mx, ebuf[b * 128 + t]);
      float sm = 0.f;
      for (int t = 0; t < L; ++t) { float e2 = __expf(ebuf[b * 128 + t] - mx); ebuf[b * 128 + t] = e2; sm += e2; }
      float inv = 1.f / sm;
      for (int t = 0; t < 128; ++t) {
        float al = (t < L) ? ebuf[b * 128 + t] * inv : 0.f;
        ebuf[b * 128 + t] = al;
        P.out_alpha[(size_t)b * 12800 + s * 128 + t] = al;
      }
    }
    __syncthreads();

    // ---- context + build aco_in = [hA, spk, ctx, ctx_s] (+ style attention)
    for (int i = tid; i < 32 * 256; i += 256) {
      int b = i >> 8, d = i & 255;
      float acc = 0.f;
      const float* ep = &ebuf[b * 128];
      const float* encp = P.enc + (size_t)(b * 128) * 256 + d;
      for (int t = 0; t < 128; ++t) acc += ep[t] * encp[t * 256];
      Xs[b * 576 + 288 + d] = (_Float16)acc;
      Xs[b * 576 + d] = hA[b * 256 + d];
    }
    for (int i = tid; i < 32 * 32; i += 256) { int b = i >> 5, d = i & 31;
      Xs[b * 576 + 256 + d] = (_Float16)P.spk_embed[P.speaker[b] * 32 + d]; }
    if (tid < 32) {
      int b = tid;
      float q0 = 0.f, q1 = 0.f;
      for (int a = 0; a < 256; ++a) {
        float h = (float)hA[b * 256 + a];
        q0 += h * P.Wq_s[a * 2];
        q1 += h * P.Wq_s[a * 2 + 1];
      }
      float es[10], mx = -1e30f;
      for (int n = 0; n < 10; ++n) {
        float v = P.v_s[0] * tanhf(P.keys_s[n * 2] + q0) +
                  P.v_s[1] * tanhf(P.keys_s[n * 2 + 1] + q1);
        es[n] = v; mx = fmaxf(mx, v);
      }
      float sm = 0.f;
      for (int n = 0; n < 10; ++n) { es[n] = __expf(es[n] - mx); sm += es[n]; }
      float c0 = 0.f, c1 = 0.f;
      for (int n = 0; n < 10; ++n) {
        float al = es[n] / sm;
        c0 += al * P.mem_style[n * 2];
        c1 += al * P.mem_style[n * 2 + 1];
      }
      Xs[b * 576 + 544] = (_Float16)c0;
      Xs[b * 576 + 545] = (_Float16)c1;
    }
    prefetch_w(P.Wat, 256 * 576 * 2);
    __syncthreads();
    wgemm(Xs, P.Wat, 576, 256, 576, P.ba, ACT_NONE, aco, 256);
    __syncthreads();

    // ---- decoder GRU1
    for (int i = tid; i < 32 * 256; i += 256) { int b = i >> 8, d = i & 255;
      Xs[b * 576 + d] = aco[b * 256 + d]; Xs[b * 576 + 256 + d] = h1s[b * 256 + d]; }
    prefetch_w(P.Wg1t, 512 * 512 * 2);
    __syncthreads();
    wgemm(Xs, P.Wg1t, 512, 512, 512, P.bg1, ACT_SIG, Gs, 512);
    __syncthreads();
    for (int i = tid; i < 32 * 256; i += 256) { int b = i >> 8, d = i & 255;
      Xs[b * 576 + 256 + d] = (_Float16)((float)Gs[b * 512 + d] * (float)h1s[b * 256 + d]); }
    prefetch_w(P.Wc1t, 256 * 512 * 2);
    __syncthreads();
    wgemm(Xs, P.Wc1t, 512, 256, 512, P.bc1, ACT_TANH, Ys, 256);
    __syncthreads();
    for (int i = tid; i < 32 * 256; i += 256) { int b = i >> 8, d = i & 255;
      float u = (float)Gs[b * 512 + 256 + d];
      float h = (float)h1s[b * 256 + d], c = (float)Ys[b * 256 + d];
      float hn = u * h + (1.f - u) * c;
      h1s[b * 256 + d] = (_Float16)hn;
      aco[b * 256 + d] = (_Float16)(hn + (float)aco[b * 256 + d]);  // y1
    }
    __syncthreads();

    // ---- decoder GRU2
    for (int i = tid; i < 32 * 256; i += 256) { int b = i >> 8, d = i & 255;
      Xs[b * 576 + d] = aco[b * 256 + d]; Xs[b * 576 + 256 + d] = h2s[b * 256 + d]; }
    prefetch_w(P.Wg2t, 512 * 512 * 2);
    __syncthreads();
    wgemm(Xs, P.Wg2t, 512, 512, 512, P.bg2, ACT_SIG, Gs, 512);
    __syncthreads();
    for (int i = tid; i < 32 * 256; i += 256) { int b = i >> 8, d = i & 255;
      Xs[b * 576 + 256 + d] = (_Float16)((float)Gs[b * 512 + d] * (float)h2s[b * 256 + d]); }
    prefetch_w(P.Wc2t, 256 * 512 * 2);
    __syncthreads();
    wgemm(Xs, P.Wc2t, 512, 256, 512, P.bc2, ACT_TANH, Ys, 256);
    __syncthreads();
    for (int i = tid; i < 32 * 256; i += 256) { int b = i >> 8, d = i & 255;
      float u = (float)Gs[b * 512 + 256 + d];
      float h = (float)h2s[b * 256 + d], c = (float)Ys[b * 256 + d];
      float hn = u * h + (1.f - u) * c;
      h2s[b * 256 + d] = (_Float16)hn;
      Xs[b * 576 + d] = (_Float16)(hn + (float)aco[b * 256 + d]);  // y2
    }
    prefetch_w(P.Wot, 640 * 256 * 2);
    __syncthreads();

    // ---- output projection Wo (N=640), store straight to global mel
    {
      const int wave = tid >> 5, lane = tid & 31;
      for (int t = wave; t < 80; t += 8) {
        const int m0 = (t & 1) << 4, n0 = (t >> 1) << 4;
        v8f c = {};
        for (int k0 = 0; k0 < 256; k0 += 32) {
          v16h a = ldfrag(Xs + m0 * 576 + k0, 576);
          v16h b = ldfrag(P.Wot + n0 * 256 + k0, 256);
          c = wmma16(a, b, c);
        }
        const int col = n0 + (lane & 15);
        const int r0  = m0 + ((lane >> 4) << 3);
        const float bv = P.bo[col];
        const int f = col >> 7, d = col & 127;
#pragma unroll
        for (int i = 0; i < 8; ++i) {
          int b = r0 + i;
          P.out_mel[(size_t)b * 64000 + (s * 5 + f) * 128 + d] = c[i] + bv;
        }
      }
    }
    __syncthreads();
  }
}

// ---------------------------------------------------------------------------
// Host launcher
// ---------------------------------------------------------------------------
extern "C" void kernel_launch(void* const* d_in, const int* in_sizes, int n_in,
                              void* d_out, int out_size, void* d_ws,
                              size_t ws_size, hipStream_t stream) {
  (void)in_sizes; (void)n_in; (void)out_size; (void)ws_size;
  const float* enc      = (const float*)d_in[0];
  const float* style_tk = (const float*)d_in[1];
  const float* mel_gt   = (const float*)d_in[2];
  const int*   speaker  = (const int*)d_in[3];
  const int*   inp_mask = (const int*)d_in[4];
  const float* spk_emb  = (const float*)d_in[5];
  const float* Wp1 = (const float*)d_in[6];  const float* bp1 = (const float*)d_in[7];
  const float* Wp2 = (const float*)d_in[8];  const float* bp2 = (const float*)d_in[9];
  const float* Wga = (const float*)d_in[10]; const float* bga = (const float*)d_in[11];
  const float* Wca = (const float*)d_in[12]; const float* bca = (const float*)d_in[13];
  const float* Wk  = (const float*)d_in[14]; const float* Wq  = (const float*)d_in[15];
  const float* v_att = (const float*)d_in[16];
  const float* Wsp = (const float*)d_in[17]; const float* bsp = (const float*)d_in[18];
  const float* Wks = (const float*)d_in[19]; const float* Wqs = (const float*)d_in[20];
  const float* v_s = (const float*)d_in[21];
  const float* Wa  = (const float*)d_in[22]; const float* ba  = (const float*)d_in[23];
  const float* Wg1 = (const float*)d_in[24]; const float* bg1 = (const float*)d_in[25];
  const float* Wc1 = (const float*)d_in[26]; const float* bc1 = (const float*)d_in[27];
  const float* Wg2 = (const float*)d_in[28]; const float* bg2 = (const float*)d_in[29];
  const float* Wc2 = (const float*)d_in[30]; const float* bc2 = (const float*)d_in[31];
  const float* Wo  = (const float*)d_in[32]; const float* bo  = (const float*)d_in[33];
  const float* Wspec = (const float*)d_in[34]; const float* bspec = (const float*)d_in[35];

  float* out_mel   = (float*)d_out;                     // 32*500*128
  float* out_spec  = out_mel + (size_t)32 * 500 * 128;  // 32*500*513
  float* out_alpha = out_spec + (size_t)32 * 500 * 513; // 32*100*128

  // workspace bump allocator
  char* wp = (char*)d_ws;
  auto alloc = [&](size_t bytes) -> char* {
    char* p = wp;
    wp += (bytes + 255) & ~(size_t)255;
    return p;
  };
  auto allocH = [&](size_t elems) { return (_Float16*)alloc(elems * 2); };
  auto allocF = [&](size_t elems) { return (float*)alloc(elems * 4); };

  _Float16* Wp1t = allocH(256 * 128);
  _Float16* Wp2t = allocH(128 * 256);
  _Float16* Wgat = allocH(512 * 416);
  _Float16* Wcat = allocH(256 * 416);
  _Float16* Wqt  = allocH(256 * 256);
  _Float16* Wat  = allocH(256 * 576);
  _Float16* Wg1t = allocH(512 * 512);
  _Float16* Wc1t = allocH(256 * 512);
  _Float16* Wg2t = allocH(512 * 512);
  _Float16* Wc2t = allocH(256 * 512);
  _Float16* Wot  = allocH(640 * 256);
  _Float16* Wkt  = allocH(256 * 256);
  _Float16* Wst  = allocH(528 * 128);   // Wspec^T padded to 528 rows
  _Float16* enc_h = allocH((size_t)4096 * 256);
  _Float16* mel_h = allocH((size_t)16000 * 128);
  float* keys      = allocF((size_t)4096 * 256);
  float* mem_style = allocF(64);
  float* keys_s    = allocF(64);

  auto conv = [&](const float* src, _Float16* dst, int K, int N, int Kpad, int Npad) {
    int total = Npad * Kpad;
    hipLaunchKernelGGL(convert_w, dim3((total + 255) / 256), dim3(256), 0, stream,
                       src, dst, K, N, Kpad, Npad);
  };
  conv(Wp1, Wp1t, 128, 256, 128, 256);
  conv(Wp2, Wp2t, 256, 128, 256, 128);
  conv(Wga, Wgat, 416, 512, 416, 512);
  conv(Wca, Wcat, 416, 256, 416, 256);
  conv(Wq,  Wqt,  256, 256, 256, 256);
  conv(Wa,  Wat,  546, 256, 576, 256);
  conv(Wg1, Wg1t, 512, 512, 512, 512);
  conv(Wc1, Wc1t, 512, 256, 512, 256);
  conv(Wg2, Wg2t, 512, 512, 512, 512);
  conv(Wc2, Wc2t, 512, 256, 512, 256);
  conv(Wo,  Wot,  256, 640, 256, 640);
  conv(Wk,  Wkt,  256, 256, 256, 256);
  conv(Wspec, Wst, 128, 513, 128, 528);

  hipLaunchKernelGGL(cast_f16, dim3((4096 * 256 + 255) / 256), dim3(256), 0,
                     stream, enc, enc_h, 4096 * 256);
  hipLaunchKernelGGL(style_pre, dim3(1), dim3(32), 0, stream, style_tk, Wsp,
                     bsp, Wks, mem_style, keys_s);

  // keys = enc @ Wk : M=4096, N=256, K=256 -> 4096 tiles / 8 waves per block
  hipLaunchKernelGGL(gemm_g, dim3(512), dim3(256), 0, stream, enc_h, 256, Wkt,
                     256, keys, 256, 256, 16, 256, (const float*)nullptr, 256);

  DecArgs A;
  A.enc = enc; A.mel = mel_gt; A.spk_embed = spk_emb; A.v_att = v_att;
  A.Wq_s = Wqs; A.v_s = v_s; A.mem_style = mem_style; A.keys_s = keys_s;
  A.keys = keys; A.speaker = speaker; A.inp_mask = inp_mask;
  A.Wp1t = Wp1t; A.Wp2t = Wp2t; A.Wgat = Wgat; A.Wcat = Wcat; A.Wqt = Wqt;
  A.Wat = Wat; A.Wg1t = Wg1t; A.Wc1t = Wc1t; A.Wg2t = Wg2t; A.Wc2t = Wc2t;
  A.Wot = Wot; A.bp1 = bp1; A.bp2 = bp2; A.bga = bga; A.bca = bca; A.ba = ba;
  A.bg1 = bg1; A.bc1 = bc1; A.bg2 = bg2; A.bc2 = bc2; A.bo = bo;
  A.out_mel = out_mel; A.out_alpha = out_alpha;
  hipLaunchKernelGGL(decoder, dim3(1), dim3(256), 0, stream, A);

  // output_spec = output_mel @ Wspec + bspec : M=16000, N=513(->528), K=128
  hipLaunchKernelGGL(cast_f16, dim3((16000 * 128 + 255) / 256), dim3(256), 0,
                     stream, out_mel, mel_h, 16000 * 128);
  hipLaunchKernelGGL(gemm_g, dim3(4125), dim3(256), 0, stream, mel_h, 128, Wst,
                     128, out_spec, 513, 1000, 33, 128, bspec, 513);
}
